// SFFusion_11235634446717
// MI455X (gfx1250) — matrix operand order repaired and checked by reading, not compiled, not run on hardware
//
#include <hip/hip_runtime.h>
#include <hip/hip_bf16.h>

// ---------------------------------------------------------------------------
// SFFusion block for MI455X (gfx1250): LN -> QKV 1x1 conv -> channel attention
// -> MLP -> 1x1 conv. GEMMs via V_WMMA_F32_16X16X32_BF16 (wave32), B operand
// staged in LDS with GLOBAL_LOAD_ASYNC_TO_LDS_B128 (double buffered).
//
// Shapes: B=8, C=512, H=W=64 (HW=4096), HN=8 heads, HD=64.
// ---------------------------------------------------------------------------

#define BB   8
#define CC   512
#define HWW  4096
#define HN_  8
#define HD_  64

typedef __bf16 bf16;
typedef __attribute__((ext_vector_type(16))) __bf16 v16bf;
typedef __attribute__((ext_vector_type(8)))  float  v8f;

union FragBF {           // 16 bf16 = 32 bytes = two 16B chunks
  uint4 u[2];
  v16bf v;
};

__device__ __forceinline__ float gelu_exact(float x) {
  return 0.5f * x * (1.0f + erff(x * 0.70710678118654752f));
}

// CDNA5 async copy: global -> LDS, 16B per lane, tracked by ASYNCcnt.
__device__ __forceinline__ void async_copy_b128(const void* g, unsigned lds_off) {
  asm volatile("global_load_async_to_lds_b128 %0, %1, off"
               :: "v"(lds_off), "v"((unsigned long long)(uintptr_t)g)
               : "memory");
}
template <int N>
__device__ __forceinline__ void s_wait_asynccnt() {
  asm volatile("s_wait_asynccnt %0" :: "n"(N) : "memory");
}

// ---------------------------------------------------------------------------
// Generic bf16 GEMM:  D[m,n] = sum_k A[m,k] * B[n,k]   (B stored [N][K])
// Block = 128 threads (4 waves). Block tile 128(M) x 64(N); each wave owns a
// 32x64 strip -> 8 WMMAs per 32-wide K-step. The 64x32 B tile is staged into
// LDS once per block via async copies (double buffered, 80B row pitch);
// A fragments are per-wave contiguous global loads.
// WMMA VGPR layouts used here (wave32; A and B share one contiguous k order):
//   A (16x32 MxK): lane l (lm=l&15, lg=l>>4): row lm, half  [k0+16lg..+15]
//   B (32x16 KxN): lane l: col lm, half  [k0+16lg..+15]
//   C (16x16 f32): vgpr i, lane l: m = 8*lg + i, n = lm
// ---------------------------------------------------------------------------
struct GemmP {
  const bf16* A;  long ldA, sAo, sAi;
  const bf16* B;  long ldB, sBo, sBi;
  int M, N, K;
  int zDiv;                 // z1 = z / zDiv, z2 = z % zDiv
  const float* bias;        // indexed by m, may be null
  int actGelu;
  const float* res; long ldR, sRo, sRi; int resT;   // resT: res[n*ldR+m]
  float* outF; long ldOF, sOFo, sOFi;               // row-major [m][n]
  bf16*  outB; long ldOB, sOBo, sOBi; int outBT;    // outBT: out[n*ldOB+m]
};

#define BROW 80            // LDS row pitch (bytes) for a 64B K-slice, 16B aligned
#define BTILE (64 * BROW)  // 5120 B per buffer

__global__ __launch_bounds__(128) void gemm_bf16_wmma(GemmP p) {
  __shared__ unsigned char smem[2 * BTILE];

  const int z  = blockIdx.z;
  const int z1 = z / p.zDiv;
  const int z2 = z % p.zDiv;
  const bf16* A = p.A + z1 * p.sAo + z2 * p.sAi;
  const bf16* B = p.B + z1 * p.sBo + z2 * p.sBi;

  const int tid  = threadIdx.x;
  const int wid  = tid >> 5;
  const int lane = tid & 31;
  const int lg   = lane >> 4;
  const int lm   = lane & 15;

  const int m0 = blockIdx.y * 128 + wid * 32;  // this wave's 32-row strip
  const int n0 = blockIdx.x * 64;

  const unsigned ldsBase = (unsigned)(uintptr_t)smem;  // low 32b = LDS offset

  // Each thread copies 2 of the 256 16B chunks of the 64x32 bf16 B tile.
  auto issueTile = [&](int buf, int k0) {
    const unsigned base = ldsBase + buf * BTILE;
#pragma unroll
    for (int j = 0; j < 2; ++j) {
      const int ci  = tid + 128 * j;       // 0..255
      const int row = ci >> 2;             // n within tile
      const int c4  = ci & 3;              // 16B chunk within 64B row
      const bf16* g = B + (long)(n0 + row) * p.ldB + k0 + c4 * 8;
      async_copy_b128(g, base + row * BROW + c4 * 16);
    }
  };

  v8f acc0[4] = {v8f{}, v8f{}, v8f{}, v8f{}};
  v8f acc1[4] = {v8f{}, v8f{}, v8f{}, v8f{}};

  const bf16* aRow0 = A + (long)(m0 + lm) * p.ldA;
  const bf16* aRow1 = A + (long)(m0 + 16 + lm) * p.ldA;

  const int nk = p.K >> 5;
  issueTile(0, 0);
  for (int kt = 0; kt < nk; ++kt) {
    const int k0 = kt << 5;
    if (kt + 1 < nk) {
      issueTile((kt + 1) & 1, k0 + 32);
      __builtin_prefetch(aRow0 + k0 + 128, 0, 0);
      __builtin_prefetch(aRow1 + k0 + 128, 0, 0);
      s_wait_asynccnt<2>();   // my 2 chunks of the current tile are done
    } else {
      s_wait_asynccnt<0>();
    }
    __syncthreads();          // everyone's chunks of the current tile are done

    FragBF fa0, fa1;
    fa0.u[0] = *(const uint4*)(aRow0 + k0 + 16 * lg);
    fa0.u[1] = *(const uint4*)(aRow0 + k0 + 16 * lg + 8);
    fa1.u[0] = *(const uint4*)(aRow1 + k0 + 16 * lg);
    fa1.u[1] = *(const uint4*)(aRow1 + k0 + 16 * lg + 8);

    const unsigned char* bufp = smem + (kt & 1) * BTILE;
#pragma unroll
    for (int t = 0; t < 4; ++t) {
      FragBF fb;
      const uint4* lp = (const uint4*)(bufp + (16 * t + lm) * BROW + lg * 32);
      fb.u[0] = lp[0];
      fb.u[1] = lp[1];
      acc0[t] = __builtin_amdgcn_wmma_f32_16x16x32_bf16(
          false, fa0.v, false, fb.v, (short)0, acc0[t], false, false);
      acc1[t] = __builtin_amdgcn_wmma_f32_16x16x32_bf16(
          false, fa1.v, false, fb.v, (short)0, acc1[t], false, false);
    }
    __syncthreads();          // all reads done before the tile is overwritten
  }

  // Epilogue
  const float* R  = p.res  ? p.res  + z1 * p.sRo  + z2 * p.sRi  : nullptr;
  float*       OF = p.outF ? p.outF + z1 * p.sOFo + z2 * p.sOFi : nullptr;
  bf16*        OB = p.outB ? p.outB + z1 * p.sOBo + z2 * p.sOBi : nullptr;
#pragma unroll
  for (int s = 0; s < 2; ++s) {
    v8f* acc = s ? acc1 : acc0;
#pragma unroll
    for (int t = 0; t < 4; ++t) {
#pragma unroll
      for (int i = 0; i < 8; ++i) {
        const int m = m0 + 16 * s + 8 * lg + i;
        const int n = n0 + 16 * t + lm;
        float v = acc[t][i];
        if (p.bias)    v += p.bias[m];
        if (p.actGelu) v  = gelu_exact(v);
        if (R)         v += p.resT ? R[(long)n * p.ldR + m] : R[(long)m * p.ldR + n];
        if (OF)        OF[(long)m * p.ldOF + n] = v;
        if (OB) {
          const long idx = p.outBT ? (long)n * p.ldOB + m : (long)m * p.ldOB + n;
          OB[idx] = (bf16)v;
        }
      }
    }
  }
}

// ---------------------------------------------------------------------------
// Attention scores + softmax per (batch, head): S = Q Kt over HW=4096,
// logits = S * invq[d] * invk[e] * HD^-1/2, row softmax, store attn^T (bf16,
// [e][d] layout so the apply-GEMM's B operand is contiguous in K).
// ---------------------------------------------------------------------------
__global__ __launch_bounds__(128) void attn_softmax_kernel(
    const bf16* __restrict__ q, const bf16* __restrict__ k,
    const float* __restrict__ invq, const float* __restrict__ invk,
    bf16* __restrict__ attnT) {
  const int z = blockIdx.x;          // b*HN + h
  const int b = z >> 3, h = z & 7;
  const bf16* Q = q + ((long)b * CC + h * HD_) * HWW;
  const bf16* K = k + ((long)b * CC + h * HD_) * HWW;

  __shared__ float S[64][65];

  const int tid  = threadIdx.x;
  const int wid  = tid >> 5;
  const int lane = tid & 31;
  const int lg   = lane >> 4;
  const int lm   = lane & 15;
  const int m0   = wid * 16;

  v8f acc[4] = {v8f{}, v8f{}, v8f{}, v8f{}};
  const bf16* aRow = Q + (long)(m0 + lm) * HWW;
  for (int k0 = 0; k0 < HWW; k0 += 32) {
    FragBF fa;
    fa.u[0] = *(const uint4*)(aRow + k0 + 16 * lg);
    fa.u[1] = *(const uint4*)(aRow + k0 + 16 * lg + 8);
#pragma unroll
    for (int t = 0; t < 4; ++t) {
      const bf16* bCol = K + (long)(16 * t + lm) * HWW + k0 + 16 * lg;
      FragBF fb;
      fb.u[0] = *(const uint4*)(bCol);
      fb.u[1] = *(const uint4*)(bCol + 8);
      acc[t] = __builtin_amdgcn_wmma_f32_16x16x32_bf16(
          false, fa.v, false, fb.v, (short)0, acc[t], false, false);
    }
  }
#pragma unroll
  for (int t = 0; t < 4; ++t)
#pragma unroll
    for (int i = 0; i < 8; ++i)
      S[m0 + 8 * lg + i][16 * t + lm] = acc[t][i];
  __syncthreads();

  if (tid < 64) {
    const int d = tid;
    const float iq = invq[(long)b * CC + h * HD_ + d];
    const float scale = 0.125f;  // HD^-0.5
    float row[64];
    float mx = -3.0e38f;
#pragma unroll 8
    for (int e = 0; e < 64; ++e) {
      const float ik = invk[(long)b * CC + h * HD_ + e];
      const float l  = S[d][e] * iq * ik * scale;
      row[e] = l;
      mx = fmaxf(mx, l);
    }
    float s = 0.f;
#pragma unroll 8
    for (int e = 0; e < 64; ++e) {
      const float pr = __expf(row[e] - mx);
      row[e] = pr;
      s += pr;
    }
    const float inv = 1.0f / s;
#pragma unroll 8
    for (int e = 0; e < 64; ++e)
      attnT[((long)z * 64 + e) * 64 + d] = (bf16)(row[e] * inv);
  }
}

// ---------------------------------------------------------------------------
// LayerNorm over the channel dim of [B,C,H,W]; writes bf16 pixel-major [HW][C]
// (B operand for projection GEMMs) and optionally f32 channel-major (residual).
// Block: 256 threads, 64 pixels.
// ---------------------------------------------------------------------------
__global__ __launch_bounds__(256) void ln_kernel(
    const float* __restrict__ x, const float* __restrict__ w,
    const float* __restrict__ bias, float* __restrict__ outF_cm,
    bf16* __restrict__ outB_pm) {
  const long blk = blockIdx.x;
  const int  b   = (int)(blk >> 6);          // HW/64 == 64 tiles/batch
  const long p0  = (blk & 63) * 64;
  const int  tid = threadIdx.x;
  const int  pp  = tid & 63;
  const int  cs  = tid >> 6;                 // 0..3
  const float* X = x + (long)b * CC * HWW;

  float s = 0.f, s2 = 0.f;
  for (int c = cs; c < CC; c += 4) {
    const float v = X[(long)c * HWW + p0 + pp];
    s += v; s2 += v * v;
  }
  __shared__ float SS[4][64], S2[4][64];
  SS[cs][pp] = s; S2[cs][pp] = s2;
  __syncthreads();
  const float ts = SS[0][pp] + SS[1][pp] + SS[2][pp] + SS[3][pp];
  const float t2 = S2[0][pp] + S2[1][pp] + S2[2][pp] + S2[3][pp];
  const float mean = ts * (1.0f / CC);
  const float var  = t2 * (1.0f / CC) - mean * mean;
  const float rstd = rsqrtf(var + 1e-5f);

  for (int c = cs; c < CC; c += 4) {
    const float v = X[(long)c * HWW + p0 + pp];
    const float o = (v - mean) * rstd * w[c] + bias[c];
    if (outF_cm) outF_cm[(long)b * CC * HWW + (long)c * HWW + p0 + pp] = o;
    outB_pm[((long)b * HWW + p0 + pp) * CC + c] = (bf16)o;
  }
}

// ---------------------------------------------------------------------------
// Row sum-of-squares -> inverse L2 norm: inv[r] = 1/max(||row||, 1e-12).
// One block per row of 4096 bf16.
// ---------------------------------------------------------------------------
__global__ __launch_bounds__(128) void rownorm_kernel(
    const bf16* __restrict__ x, float* __restrict__ invn) {
  const long r = blockIdx.x;
  const bf16* row = x + r * (long)HWW;
  float s = 0.f;
  for (int i = threadIdx.x; i < HWW; i += 128) {
    const float v = (float)row[i];
    s += v * v;
  }
  __shared__ float red[128];
  red[threadIdx.x] = s;
  __syncthreads();
  for (int st = 64; st > 0; st >>= 1) {
    if (threadIdx.x < st) red[threadIdx.x] += red[threadIdx.x + st];
    __syncthreads();
  }
  if (threadIdx.x == 0) invn[r] = 1.0f / fmaxf(sqrtf(red[0]), 1e-12f);
}

// ---------------------------------------------------------------------------
// Weight f32 -> bf16, optional transpose. dst row-major [R][Cc];
// trans: src is [Cc][R].
// ---------------------------------------------------------------------------
__global__ __launch_bounds__(256) void wconv_kernel(
    const float* __restrict__ src, bf16* __restrict__ dst,
    int R, int Cc, int trans) {
  const long i = (long)blockIdx.x * 256 + threadIdx.x;
  if (i >= (long)R * Cc) return;
  const int r = (int)(i / Cc), c = (int)(i % Cc);
  const float v = trans ? src[(long)c * R + r] : src[i];
  dst[i] = (bf16)v;
}

// ---------------------------------------------------------------------------
// Host-side orchestration
// ---------------------------------------------------------------------------
static inline GemmP zeroGemm() { GemmP p; __builtin_memset(&p, 0, sizeof(p)); p.zDiv = 1; return p; }

extern "C" void kernel_launch(void* const* d_in, const int* in_sizes, int n_in,
                              void* d_out, int out_size, void* d_ws, size_t ws_size,
                              hipStream_t stream) {
  (void)in_sizes; (void)n_in; (void)out_size; (void)ws_size;

  const float* x_opt = (const float*)d_in[0];
  const float* x_sar = (const float*)d_in[1];
  const float* ln_o_w = (const float*)d_in[2];
  const float* ln_o_b = (const float*)d_in[3];
  const float* ln_s_w = (const float*)d_in[4];
  const float* ln_s_b = (const float*)d_in[5];
  const float* wq = (const float*)d_in[6];  const float* bq = (const float*)d_in[7];
  const float* wk = (const float*)d_in[8];  const float* bk = (const float*)d_in[9];
  const float* wv = (const float*)d_in[10]; const float* bv = (const float*)d_in[11];
  const float* w1 = (const float*)d_in[12]; const float* b1 = (const float*)d_in[13];
  const float* w2 = (const float*)d_in[14]; const float* b2 = (const float*)d_in[15];
  const float* wo = (const float*)d_in[16]; const float* bo = (const float*)d_in[17];
  float* out = (float*)d_out;

  // ---- workspace layout (with aliasing of dead buffers) ----
  char* ws = (char*)d_ws;
  size_t off = 0;
  auto alloc = [&](size_t bytes) -> void* {
    void* p = ws + off;
    off += (bytes + 255) & ~(size_t)255;
    return p;
  };
  const size_t E = (size_t)BB * CC * HWW;           // 16.78M elements
  float* xo_f32  = (float*)alloc(E * 4);            // LN(x_opt), channel-major
  bf16*  xo_t_bf = (bf16*)alloc(E * 2);             // LN(x_opt), pixel-major
  bf16*  xs_t_bf = (bf16*)alloc(E * 2);             // LN(x_sar), pixel-major
  bf16*  q_bf    = (bf16*)alloc(E * 2);             // q, channel-major [C][HW]
  bf16*  k_bf    = (bf16*)alloc(E * 2);             // k, channel-major [C][HW]
  bf16*  v_t     = (bf16*)alloc(E * 2);             // v, pixel-major [HW][C]
  float* u_f32   = (float*)alloc(E * 4);            // z_sum f32, pixel-major
  bf16*  attnT   = (bf16*)alloc((size_t)BB * HN_ * 64 * 64 * 2);
  float* invq    = (float*)alloc((size_t)BB * CC * 4);
  float* invk    = (float*)alloc((size_t)BB * CC * 4);
  bf16*  wq_bf   = (bf16*)alloc((size_t)CC * CC * 2);
  bf16*  wk_bf   = (bf16*)alloc((size_t)CC * CC * 2);
  bf16*  wv_bf   = (bf16*)alloc((size_t)CC * CC * 2);
  bf16*  wo_bf   = (bf16*)alloc((size_t)CC * CC * 2);
  bf16*  w1t_bf  = (bf16*)alloc((size_t)2 * CC * CC * 2);   // [2C][C]
  bf16*  w2t_bf  = (bf16*)alloc((size_t)CC * 2 * CC * 2);   // [C][2C]
  // Aliases (safe by dataflow ordering):
  bf16*  u_bf  = xs_t_bf;   // z_sum bf16 pixel-major (xs dead after k,v proj)
  bf16*  h_bf  = q_bf;      // MLP hidden [HW][2C] over q_bf+k_bf (dead after attn)
  bf16*  u2_bf = xo_t_bf;   // post-MLP pixel-major (xo_t dead after q proj)

  // ---- 1) weights -> bf16 (w1, w2 transposed so A operands are [M][K]) ----
  {
    const int n512 = (CC * CC + 255) / 256;        // 1024 blocks
    const int n1024 = (2 * CC * CC + 255) / 256;
    wconv_kernel<<<n512, 256, 0, stream>>>(wq, wq_bf, CC, CC, 0);
    wconv_kernel<<<n512, 256, 0, stream>>>(wk, wk_bf, CC, CC, 0);
    wconv_kernel<<<n512, 256, 0, stream>>>(wv, wv_bf, CC, CC, 0);
    wconv_kernel<<<n512, 256, 0, stream>>>(wo, wo_bf, CC, CC, 0);
    wconv_kernel<<<n1024, 256, 0, stream>>>(w1, w1t_bf, 2 * CC, CC, 1); // w1t[j][c]=w1[c][j]
    wconv_kernel<<<n1024, 256, 0, stream>>>(w2, w2t_bf, CC, 2 * CC, 1); // w2t[d][j]=w2[j][d]
  }

  // ---- 2) LayerNorms ----
  ln_kernel<<<BB * 64, 256, 0, stream>>>(x_opt, ln_o_w, ln_o_b, xo_f32, xo_t_bf);
  ln_kernel<<<BB * 64, 256, 0, stream>>>(x_sar, ln_s_w, ln_s_b, nullptr, xs_t_bf);

  const long EC = (long)HWW * CC;   // per-batch activation elements
  const dim3 blk128(128);

  // ---- 3) Q/K/V projections (M=512, N=4096, K=512 per batch) ----
  {
    GemmP p = zeroGemm();
    p.ldA = CC; p.ldB = CC; p.sBo = EC;
    p.M = CC; p.N = HWW; p.K = CC;
    // Q: out bf16 channel-major [C][HW]
    p.A = wq_bf; p.B = xo_t_bf; p.bias = bq;
    p.outB = q_bf; p.ldOB = HWW; p.sOBo = EC; p.outBT = 0;
    gemm_bf16_wmma<<<dim3(HWW / 64, CC / 128, BB), blk128, 0, stream>>>(p);
    // K
    p.A = wk_bf; p.B = xs_t_bf; p.bias = bk; p.outB = k_bf;
    gemm_bf16_wmma<<<dim3(HWW / 64, CC / 128, BB), blk128, 0, stream>>>(p);
    // V: store transposed pixel-major [HW][C]
    p.A = wv_bf; p.bias = bv; p.outB = v_t; p.ldOB = CC; p.outBT = 1;
    gemm_bf16_wmma<<<dim3(HWW / 64, CC / 128, BB), blk128, 0, stream>>>(p);
  }

  // ---- 4) inverse L2 norms of q,k rows (over HW) ----
  rownorm_kernel<<<BB * CC, 128, 0, stream>>>(q_bf, invq);
  rownorm_kernel<<<BB * CC, 128, 0, stream>>>(k_bf, invk);

  // ---- 5) attention scores + softmax ----
  attn_softmax_kernel<<<BB * HN_, blk128, 0, stream>>>(q_bf, k_bf, invq, invk, attnT);

  // ---- 6) apply attention: z[p, he+e] = sum_d v_t[p, hd+d] * attnT[e][d];
  //         add x_opt residual (z_opt = xo^T) -> u (f32 + bf16, pixel-major)
  {
    GemmP p = zeroGemm();
    p.zDiv = HN_;                          // z = b*8 + h
    p.A = v_t;   p.ldA = CC;  p.sAo = EC; p.sAi = HD_;
    p.B = attnT; p.ldB = HD_; p.sBo = (long)HN_ * 64 * 64; p.sBi = 64 * 64;
    p.M = HWW; p.N = HD_; p.K = HD_;
    p.res = xo_f32; p.ldR = HWW; p.sRo = EC; p.sRi = (long)HD_ * HWW; p.resT = 1;
    p.outF = u_f32; p.ldOF = CC; p.sOFo = EC; p.sOFi = HD_;
    p.outB = u_bf;  p.ldOB = CC; p.sOBo = EC; p.sOBi = HD_; p.outBT = 0;
    gemm_bf16_wmma<<<dim3(1, HWW / 128, BB * HN_), blk128, 0, stream>>>(p);
  }

  // ---- 7) MLP layer 1: h = gelu(u @ w1 + b1), stored pixel-major [HW][2C] ----
  {
    GemmP p = zeroGemm();
    p.A = w1t_bf; p.ldA = CC;
    p.B = u_bf;   p.ldB = CC; p.sBo = EC;
    p.M = 2 * CC; p.N = HWW; p.K = CC;
    p.bias = b1; p.actGelu = 1;
    p.outB = h_bf; p.ldOB = 2 * CC; p.sOBo = 2 * EC; p.outBT = 1;
    gemm_bf16_wmma<<<dim3(HWW / 64, 2 * CC / 128, BB), blk128, 0, stream>>>(p);
  }

  // ---- 8) MLP layer 2 + residual: u2 = h @ w2 + b2 + u ----
  {
    GemmP p = zeroGemm();
    p.A = w2t_bf; p.ldA = 2 * CC;
    p.B = h_bf;   p.ldB = 2 * CC; p.sBo = 2 * EC;
    p.M = CC; p.N = HWW; p.K = 2 * CC;
    p.bias = b2;
    p.res = u_f32; p.ldR = CC; p.sRo = EC; p.resT = 1;
    p.outB = u2_bf; p.ldOB = CC; p.sOBo = EC; p.outBT = 1;
    gemm_bf16_wmma<<<dim3(HWW / 64, CC / 128, BB), blk128, 0, stream>>>(p);
  }

  // ---- 9) output conv + final residual: y = xo + wo @ u2 + bo ----
  {
    GemmP p = zeroGemm();
    p.A = wo_bf; p.ldA = CC;
    p.B = u2_bf; p.ldB = CC; p.sBo = EC;
    p.M = CC; p.N = HWW; p.K = CC;
    p.bias = bo;
    p.res = xo_f32; p.ldR = HWW; p.sRo = EC; p.resT = 0;
    p.outF = out; p.ldOF = HWW; p.sOFo = EC;
    gemm_bf16_wmma<<<dim3(HWW / 64, CC / 128, BB), blk128, 0, stream>>>(p);
  }
}
